// GroupwiseMMD_37529424232489
// MI455X (gfx1250) — matrix operand order repaired and checked by reading, not compiled, not run on hardware
//
#include <hip/hip_runtime.h>
#include <cstdint>
#include <cstddef>

// ---------------- types ----------------
typedef __attribute__((ext_vector_type(16))) __bf16 v16bf;
typedef __attribute__((ext_vector_type(8)))  float  v8f;
typedef __attribute__((ext_vector_type(4)))  float  f32x4;
typedef __attribute__((ext_vector_type(4)))  __bf16 bf16x4;

#define TT 128          // square tile edge per workgroup
#define TK 32           // K step (bf16 wmma depth)
#define NTHREADS 256    // 8 waves (wave32)
#define BSTRIDE 40      // padded bf16 row stride in elements (80 bytes, 16B aligned)
#define MT 2            // 16-row tiles per wave (32 rows)
#define NT 4            // 16-col tiles per wave (64 cols)

#ifndef __has_builtin
#define __has_builtin(x) 0
#endif

#if __has_builtin(__builtin_amdgcn_global_load_async_to_lds_b128)
#define USE_ASYNC_LDS 1
typedef int v4i __attribute__((vector_size(16)));
typedef __attribute__((address_space(1))) v4i* gv4p;
typedef __attribute__((address_space(3))) v4i* lv4p;
#define ASYNC_CP16(gp, lp) \
    __builtin_amdgcn_global_load_async_to_lds_b128((gv4p)(gp), (lv4p)(lp), 0, 0)
#else
#define USE_ASYNC_LDS 0
#endif

union FragU { f32x4 q[2]; v16bf v; };

__device__ __forceinline__ void waitAsync() {
#if USE_ASYNC_LDS
#if __has_builtin(__builtin_amdgcn_s_wait_asynccnt)
    __builtin_amdgcn_s_wait_asynccnt(0);
#else
    asm volatile("s_wait_asynccnt 0" ::: "memory");
#endif
#endif
}

// ---------------- kernel 1: row norms + weights ----------------
__global__ __launch_bounds__(256) void GroupwiseMMD_prep(
    const float* __restrict__ z, const int* __restrict__ c,
    float* __restrict__ norms, float* __restrict__ w0f, float* __restrict__ w1f,
    int n, int d)
{
    const int row = blockIdx.x;
    float s = 0.f;
    for (int j = threadIdx.x; j < d; j += 256) {
        float v = z[(size_t)row * d + j];
        s = fmaf(v, v, s);
    }
    for (int off = 16; off; off >>= 1) s += __shfl_xor(s, off, 32);
    __shared__ float buf[8];
    const int lane = threadIdx.x & 31, wave = threadIdx.x >> 5;
    if (lane == 0) buf[wave] = s;
    __syncthreads();
    if (threadIdx.x == 0) {
        float tot = 0.f;
        for (int w = 0; w < 8; ++w) tot += buf[w];
        norms[row] = tot;
        w0f[row] = (float)c[2 * row + 0];
        w1f[row] = (float)c[2 * row + 1];
    }
}

// ---------------- kernel 2: fused triangular tile GEMM + RBF + reduction ----------------
// Exploits K symmetry: only tile pairs (I<=J) are computed; off-diagonal tiles
// contribute twice (s00/s11 doubled, s01 symmetrized: a0*w1col + a1*w0col).
__global__ __launch_bounds__(NTHREADS) void GroupwiseMMD_tile(
    const float* __restrict__ z,
    const float* __restrict__ norms,
    const float* __restrict__ w0f, const float* __restrict__ w1f,
    float* __restrict__ partials, int d, int T)
{
    __shared__ __align__(16) float  rawA[TT * TK];
    __shared__ __align__(16) float  rawB[TT * TK];
    __shared__ __align__(16) __bf16 Ahi[TT * BSTRIDE];
    __shared__ __align__(16) __bf16 Alo[TT * BSTRIDE];
    __shared__ __align__(16) __bf16 Bhi[TT * BSTRIDE];
    __shared__ __align__(16) __bf16 Blo[TT * BSTRIDE];
    __shared__ float redbuf[8 * 3];

    // decode triangular pair index (uniform SALU loop, <= T iters)
    int I = 0, rem = blockIdx.x;
    while (rem >= T - I) { rem -= T - I; ++I; }
    const int J = I + rem;

    const int t    = threadIdx.x;
    const int lane = t & 31;
    const int wave = t >> 5;
    const int wm   = wave >> 1;   // 0..3  (32-row groups)
    const int wn   = wave & 1;    // 0..1  (64-col groups)
    const int half = lane >> 4;   // lane half (wave32 WMMA layout)
    const int lr   = lane & 15;

    const int tileM = I * TT;
    const int tileN = J * TT;

    v8f acc[MT][NT] = {};

    for (int k0 = 0; k0 < d; k0 += TK) {
        // -------- stage raw f32 tiles into LDS (async copy path) --------
        for (int i = 0; i < 4; ++i) {            // A: 1024 x 16B chunks
            int id = t + i * NTHREADS;
            int r = id >> 3, cc = id & 7;
            const float* gp = z + (size_t)(tileM + r) * d + k0 + cc * 4;
            float* lp = rawA + r * TK + cc * 4;
#if USE_ASYNC_LDS
            ASYNC_CP16(gp, lp);
#else
            *(f32x4*)lp = *(const f32x4*)gp;
#endif
        }
        for (int i = 0; i < 4; ++i) {            // B: 1024 x 16B chunks
            int id = t + i * NTHREADS;
            int r = id >> 3, cc = id & 7;
            const float* gp = z + (size_t)(tileN + r) * d + k0 + cc * 4;
            float* lp = rawB + r * TK + cc * 4;
#if USE_ASYNC_LDS
            ASYNC_CP16(gp, lp);
#else
            *(f32x4*)lp = *(const f32x4*)gp;
#endif
        }
        waitAsync();
        __syncthreads();

        // -------- split f32 -> bf16 hi/lo into padded LDS --------
        for (int i = 0; i < 4; ++i) {
            int id = t + i * NTHREADS;
            int r = id >> 3, cc = id & 7;
            f32x4 v = *(const f32x4*)(rawA + r * TK + cc * 4);
            bf16x4 h, l;
            for (int j = 0; j < 4; ++j) {
                h[j] = (__bf16)v[j];
                l[j] = (__bf16)(v[j] - (float)h[j]);
            }
            *(bf16x4*)(Ahi + r * BSTRIDE + cc * 4) = h;
            *(bf16x4*)(Alo + r * BSTRIDE + cc * 4) = l;
        }
        for (int i = 0; i < 4; ++i) {
            int id = t + i * NTHREADS;
            int r = id >> 3, cc = id & 7;
            f32x4 v = *(const f32x4*)(rawB + r * TK + cc * 4);
            bf16x4 h, l;
            for (int j = 0; j < 4; ++j) {
                h[j] = (__bf16)v[j];
                l[j] = (__bf16)(v[j] - (float)h[j]);
            }
            *(bf16x4*)(Bhi + r * BSTRIDE + cc * 4) = h;
            *(bf16x4*)(Blo + r * BSTRIDE + cc * 4) = l;
        }
        __syncthreads();

        // -------- fragments (wave32 WMMA layouts) & multiply --------
        FragU aH[MT], aL[MT];
        for (int mt = 0; mt < MT; ++mt) {
            int r = wm * 32 + mt * 16 + lr;
            // A 16x32 bf16: lanes0-15 -> K{0..7,16..23}, lanes16-31 -> K{8..15,24..31}
            const __bf16* bh = Ahi + r * BSTRIDE + half * 8;
            const __bf16* bl = Alo + r * BSTRIDE + half * 8;
            aH[mt].q[0] = *(const f32x4*)(bh);
            aH[mt].q[1] = *(const f32x4*)(bh + 16);
            aL[mt].q[0] = *(const f32x4*)(bl);
            aL[mt].q[1] = *(const f32x4*)(bl + 16);
        }
        for (int nt = 0; nt < NT; ++nt) {
            int r = wn * 64 + nt * 16 + lr;
            // B 32x16 bf16: lanes0-15 -> K{0..7,16..23}, lanes16-31 -> K{8..15,24..31}
            const __bf16* bhp = Bhi + r * BSTRIDE + half * 8;
            const __bf16* blp = Blo + r * BSTRIDE + half * 8;
            FragU bH, bL;
            bH.q[0] = *(const f32x4*)(bhp);
            bH.q[1] = *(const f32x4*)(bhp + 16);
            bL.q[0] = *(const f32x4*)(blp);
            bL.q[1] = *(const f32x4*)(blp + 16);
            for (int mt = 0; mt < MT; ++mt) {
                acc[mt][nt] = __builtin_amdgcn_wmma_f32_16x16x32_bf16(
                    false, aH[mt].v, false, bH.v, (short)0, acc[mt][nt], false, false);
                acc[mt][nt] = __builtin_amdgcn_wmma_f32_16x16x32_bf16(
                    false, aH[mt].v, false, bL.v, (short)0, acc[mt][nt], false, false);
                acc[mt][nt] = __builtin_amdgcn_wmma_f32_16x16x32_bf16(
                    false, aL[mt].v, false, bH.v, (short)0, acc[mt][nt], false, false);
            }
        }
        __syncthreads();   // before next k-step overwrites LDS
    }

    // -------- epilogue: dist -> exp -> weighted partial sums (symmetry-aware) --------
    const float f  = (I == J) ? 1.f : 2.f;   // off-diagonal tiles count twice
    const float g1 = f - 1.f;                // transpose-block weight for s01
    float p00 = 0.f, p01 = 0.f, p11 = 0.f;
    for (int mt = 0; mt < MT; ++mt) {
        // C layout: VGPR r holds M = r + 8*half, N = lane%16
        const int rbase = tileM + wm * 32 + mt * 16 + (half << 3);
        float rn[8], rw0[8], rw1[8];
        for (int r = 0; r < 8; ++r) {
            rn[r]  = norms[rbase + r];
            rw0[r] = w0f[rbase + r];
            rw1[r] = w1f[rbase + r];
        }
        for (int nt = 0; nt < NT; ++nt) {
            const int gc = tileN + wn * 64 + nt * 16 + lr;
            const float cn = norms[gc], cw0 = w0f[gc], cw1 = w1f[gc];
            float a0 = 0.f, a1 = 0.f;
            for (int r = 0; r < 8; ++r) {
                float g  = acc[mt][nt][r];
                float sq = fmaxf(rn[r] + cn - 2.f * g, 0.f);
                float kv = __expf(-0.5f * sqrtf(sq));
                a0 = fmaf(kv, rw0[r], a0);
                a1 = fmaf(kv, rw1[r], a1);
            }
            p00 = fmaf(f * a0, cw0, p00);
            p11 = fmaf(f * a1, cw1, p11);
            p01 = fmaf(a0, cw1, p01);
            p01 = fmaf(g1 * a1, cw0, p01);
        }
    }
    for (int off = 16; off; off >>= 1) {
        p00 += __shfl_xor(p00, off, 32);
        p01 += __shfl_xor(p01, off, 32);
        p11 += __shfl_xor(p11, off, 32);
    }
    if (lane == 0) {
        redbuf[wave * 3 + 0] = p00;
        redbuf[wave * 3 + 1] = p01;
        redbuf[wave * 3 + 2] = p11;
    }
    __syncthreads();
    if (t == 0) {
        float s00 = 0.f, s01 = 0.f, s11 = 0.f;
        for (int w = 0; w < 8; ++w) {
            s00 += redbuf[w * 3 + 0];
            s01 += redbuf[w * 3 + 1];
            s11 += redbuf[w * 3 + 2];
        }
        partials[blockIdx.x * 3 + 0] = s00;
        partials[blockIdx.x * 3 + 1] = s01;
        partials[blockIdx.x * 3 + 2] = s11;
    }
}

// ---------------- kernel 3: deterministic final reduction ----------------
__device__ __forceinline__ float blockSum256(float v, float* buf) {
    for (int off = 16; off; off >>= 1) v += __shfl_xor(v, off, 32);
    const int lane = threadIdx.x & 31, wave = threadIdx.x >> 5;
    if (lane == 0) buf[wave] = v;
    __syncthreads();
    float tot = 0.f;
    if (threadIdx.x == 0)
        for (int w = 0; w < 8; ++w) tot += buf[w];
    __syncthreads();
    return tot;  // valid on thread 0
}

__global__ __launch_bounds__(256) void GroupwiseMMD_final(
    const int* __restrict__ c, const float* __restrict__ partials,
    int n, int ntiles, float* __restrict__ out)
{
    __shared__ float buf[8];
    float s0 = 0.f, s1 = 0.f, s00 = 0.f, s01 = 0.f, s11 = 0.f;
    for (int i = threadIdx.x; i < n; i += 256) {
        s0 += (float)c[2 * i + 0];
        s1 += (float)c[2 * i + 1];
    }
    for (int i = threadIdx.x; i < ntiles; i += 256) {
        s00 += partials[i * 3 + 0];
        s01 += partials[i * 3 + 1];
        s11 += partials[i * 3 + 2];
    }
    float t0  = blockSum256(s0,  buf);
    float t1  = blockSum256(s1,  buf);
    float t00 = blockSum256(s00, buf);
    float t01 = blockSum256(s01, buf);
    float t11 = blockSum256(s11, buf);
    if (threadIdx.x == 0) {
        float m00 = t00 / (t0 * t0);
        float m01 = t01 / (t0 * t1);
        float m11 = t11 / (t1 * t1);
        out[0] = m00 - 2.f * m01 + m11;
    }
}

// ---------------- host launcher ----------------
extern "C" void kernel_launch(void* const* d_in, const int* in_sizes, int n_in,
                              void* d_out, int out_size, void* d_ws, size_t ws_size,
                              hipStream_t stream) {
    const int*   c = (const int*)d_in[0];
    const float* z = (const float*)d_in[1];
    const int n = in_sizes[0] / 2;          // 8192
    const int d = in_sizes[1] / n;          // 256

    float* norms    = (float*)d_ws;
    float* w0f      = norms + n;
    float* w1f      = w0f + n;
    float* partials = w1f + n;

    const int T = n / TT;                   // 64 tile rows/cols
    const int npairs = T * (T + 1) / 2;     // 2080 triangular tiles

    GroupwiseMMD_prep<<<n, 256, 0, stream>>>(z, c, norms, w0f, w1f, n, d);
    GroupwiseMMD_tile<<<npairs, NTHREADS, 0, stream>>>(z, norms, w0f, w1f, partials, d, T);
    GroupwiseMMD_final<<<1, 256, 0, stream>>>(c, partials, n, npairs, (float*)d_out);
}
